// CausalModulatedAttention_4767413698784
// MI455X (gfx1250) — matrix operand order repaired and checked by reading, not compiled, not run on hardware
//
#include <hip/hip_runtime.h>
#include <hip/hip_bf16.h>
#include <cstdint>

// ---------------------------------------------------------------------------
// CausalModulatedAttention for MI455X (gfx1250, wave32, WMMA bf16 16x16x32)
// B=4, L=512, D=1024, H=16, HD=64, CD=32, ALPHA=0.3
// ---------------------------------------------------------------------------

typedef __attribute__((ext_vector_type(16))) __bf16 v16bf;
typedef __attribute__((ext_vector_type(8)))  float  v8f;

#define WMMA_BF16(a, b, c) \
    __builtin_amdgcn_wmma_f32_16x16x32_bf16(false, (a), false, (b), (short)0, (c), false, false)

static constexpr int Bsz = 4, Lsz = 512, Dsz = 1024, Hn = 16, HD = 64, CD = 32;
static constexpr float ALPHA = 0.3f;
static constexpr float NEG_BIG = 1e30f;

// ---- CDNA5 async global->LDS copy (ASYNCcnt path, ISA 10.7/15.18) ----------
// LDS destination offset = low 32 bits of the generic shared-memory pointer
// (LDS aperture: addr[31:0] is the LDS byte offset).

__device__ __forceinline__ void async_copy_b128(void* lds_dst, const void* gsrc) {
    uint32_t lds_off = (uint32_t)(uintptr_t)lds_dst;
    uint64_t gaddr   = (uint64_t)(uintptr_t)gsrc;
    asm volatile("global_load_async_to_lds_b128 %0, %1, off"
                 :: "v"(lds_off), "v"(gaddr)
                 : "memory");
}
__device__ __forceinline__ void wait_async0() {
    asm volatile("s_wait_asynccnt 0x0" ::: "memory");
}

// ---- WMMA fragment loaders (contiguous k = 16*half + e order, wave32) ----------

// A fragment: 16x32 bf16 tile of row-major matrix A (lda = row stride).
// lane m = lane%16 supplies row m; element e -> K = 16*half + e
__device__ __forceinline__ v16bf load_a_rm(const __bf16* A, int row0, int k0, int lda) {
    const int lane = threadIdx.x & 31;
    const int m = lane & 15, half = lane >> 4;
    const __bf16* p = A + (size_t)(row0 + m) * lda + k0;
    v16bf a;
#pragma unroll
    for (int e = 0; e < 16; ++e) {
        int k = 16 * half + e;
        a[e] = p[k];
    }
    return a;
}

// B fragment: 32x16 bf16 tile, logical B[k][n] fetched from Bt laid out [n][k]
// (ld = K stride): element e of lane (n=lane%16, half) is k = 16*half + e.
__device__ __forceinline__ v16bf load_b_nt(const __bf16* Bt, int n0, int k0, int ld) {
    const int lane = threadIdx.x & 31;
    const int n = lane & 15, half = lane >> 4;
    const __bf16* p = Bt + (size_t)(n0 + n) * ld + k0 + 16 * half;
    v16bf b;
#pragma unroll
    for (int e = 0; e < 16; ++e) b[e] = p[e];
    return b;
}

// B fragment from a row-major B[k][n] matrix (ld = row stride): k = 16*half + e.
__device__ __forceinline__ v16bf load_b_rm(const __bf16* Bm, int k0, int n0, int ld) {
    const int lane = threadIdx.x & 31;
    const int n = lane & 15, half = lane >> 4;
    const __bf16* p = Bm + (size_t)(k0 + 16 * half) * ld + n0 + n;
    v16bf b;
#pragma unroll
    for (int e = 0; e < 16; ++e) b[e] = p[(size_t)e * ld];
    return b;
}

// ---- Conversions -----------------------------------------------------------

__global__ void __launch_bounds__(256) to_bf16_kernel(const float* __restrict__ in,
                                                      __bf16* __restrict__ out, int n) {
    int i = blockIdx.x * blockDim.x + threadIdx.x;
    if (i < n) out[i] = (__bf16)in[i];
}

// W: K x N row-major (in,out) -> Wt: N x K row-major bf16
__global__ void __launch_bounds__(256) transpose_bf16_kernel(const float* __restrict__ W,
                                                             __bf16* __restrict__ Wt,
                                                             int K, int N) {
    int idx = blockIdx.x * blockDim.x + threadIdx.x;
    if (idx >= K * N) return;
    int n = idx / K, k = idx % K;
    Wt[(size_t)n * K + k] = (__bf16)W[(size_t)k * N + n];
}

// ---- Blocked WMMA GEMM: C(MxN) = A(MxK,bf16) @ B via Bt[N][K] --------------
// Block = 4 waves, block tile 32(M) x 128(N); each wave computes a 2x2 grid of
// 16x16 tiles (4 WMMAs per K-chunk from 2 A-frags + 2 B-frags). The shared
// 32x32 A chunk is staged to LDS with double-buffered async copies (ASYNCcnt).

__global__ void __launch_bounds__(128)
gemm_bf16_wmma_blocked(const __bf16* __restrict__ A, const __bf16* __restrict__ Bt,
                       void* __restrict__ Cout, int M, int N, int K, int store_bf16) {
    __shared__ __bf16 sA[2][32][32];                // 2 x 2 KB double buffer

    const int tid  = threadIdx.x;
    const int lane = tid & 31;
    const int wave = tid >> 5;
    const int m0 = blockIdx.y * 32;
    const int n0 = blockIdx.x * 128 + wave * 32;

    // A staging: thread t copies 16 B: row = t>>2, byte segment (t&3)*16
    const int arow = tid >> 2;
    const int acol = (tid & 3) * 8;                 // in bf16 elements

    async_copy_b128(&sA[0][arow][acol], A + (size_t)(m0 + arow) * K + acol);
    wait_async0();
    __syncthreads();

    v8f acc00 = {}, acc01 = {}, acc10 = {}, acc11 = {};
    for (int k0 = 0; k0 < K; k0 += 32) {
        const int buf = (k0 >> 5) & 1;
        const bool more = (k0 + 32) < K;
        if (more) {
            async_copy_b128(&sA[buf ^ 1][arow][acol],
                            A + (size_t)(m0 + arow) * K + k0 + 32 + acol);
            __builtin_prefetch(Bt + (size_t)(n0 + (lane & 15)) * K + k0 + 32, 0, 1);
        }
        const __bf16* As = &sA[buf][0][0];
        v16bf a0 = load_a_rm(As, 0,  0, 32);        // rows m0..m0+15  (LDS)
        v16bf a1 = load_a_rm(As, 16, 0, 32);        // rows m0+16..+31 (LDS)
        v16bf b0 = load_b_nt(Bt, n0,      k0, K);
        v16bf b1 = load_b_nt(Bt, n0 + 16, k0, K);
        acc00 = WMMA_BF16(a0, b0, acc00);
        acc01 = WMMA_BF16(a0, b1, acc01);
        acc10 = WMMA_BF16(a1, b0, acc10);
        acc11 = WMMA_BF16(a1, b1, acc11);
        if (more) wait_async0();
        __syncthreads();
    }

    const int half = lane >> 4, n = lane & 15;
    if (store_bf16) {
        __bf16* C = (__bf16*)Cout;
#pragma unroll
        for (int r = 0; r < 8; ++r) {
            const int mr0 = m0 + r + 8 * half, mr1 = mr0 + 16;
            C[(size_t)mr0 * N + n0 + n]      = (__bf16)acc00[r];
            C[(size_t)mr0 * N + n0 + 16 + n] = (__bf16)acc01[r];
            C[(size_t)mr1 * N + n0 + n]      = (__bf16)acc10[r];
            C[(size_t)mr1 * N + n0 + 16 + n] = (__bf16)acc11[r];
        }
    } else {
        float* C = (float*)Cout;
#pragma unroll
        for (int r = 0; r < 8; ++r) {
            const int mr0 = m0 + r + 8 * half, mr1 = mr0 + 16;
            C[(size_t)mr0 * N + n0 + n]      = acc00[r];
            C[(size_t)mr0 * N + n0 + 16 + n] = acc01[r];
            C[(size_t)mr1 * N + n0 + n]      = acc10[r];
            C[(size_t)mr1 * N + n0 + 16 + n] = acc11[r];
        }
    }
}

// ---- Simple WMMA GEMM (for narrow N, e.g. N=32 cause/effect) ---------------

__global__ void __launch_bounds__(128)
gemm_bf16_wmma(const __bf16* __restrict__ A, const __bf16* __restrict__ Bt,
               void* __restrict__ Cout, int M, int N, int K, int store_bf16) {
    const int lane = threadIdx.x & 31;
    const int wave = threadIdx.x >> 5;
    const int n0 = (blockIdx.x * 4 + wave) * 16;
    const int m0 = blockIdx.y * 16;
    if (n0 >= N) return;                      // uniform per wave: EXEC stays all-1s

    v8f acc = {};
    for (int k0 = 0; k0 < K; k0 += 32) {
        if (k0 + 32 < K)
            __builtin_prefetch(A + (size_t)(m0 + (lane & 15)) * K + k0 + 32, 0, 1);
        v16bf a = load_a_rm(A, m0, k0, K);
        v16bf b = load_b_nt(Bt, n0, k0, K);
        acc = WMMA_BF16(a, b, acc);
    }

    const int half = lane >> 4, n = lane & 15;
    if (store_bf16) {
        __bf16* C = (__bf16*)Cout;
#pragma unroll
        for (int r = 0; r < 8; ++r)
            C[(size_t)(m0 + r + 8 * half) * N + n0 + n] = (__bf16)acc[r];
    } else {
        float* C = (float*)Cout;
#pragma unroll
        for (int r = 0; r < 8; ++r)
            C[(size_t)(m0 + r + 8 * half) * N + n0 + n] = acc[r];
    }
}

// ---- Tiny fp32 MLP projection: out[i][c] = bias?b1[c]:0 + sum_k in[i][k]*W1[k][c]

__global__ void __launch_bounds__(256)
mlp_proj_kernel(const float* __restrict__ in, const float* __restrict__ W1,
                const float* __restrict__ b1, float* __restrict__ out, int add_bias) {
    int idx = blockIdx.x * blockDim.x + threadIdx.x;     // Bsz*Lsz*CD threads
    if (idx >= Bsz * Lsz * CD) return;
    int i = idx >> 5, c = idx & 31;
    const float* row = in + (size_t)i * CD;
    float acc = add_bias ? b1[c] : 0.0f;
#pragma unroll
    for (int k = 0; k < CD; ++k) acc += row[k] * W1[k * CD + c];
    out[idx] = acc;
}

// ---- Pairwise causal-graph bias: Gmod[b,i,j] = ALPHA*(sigmoid(gelu(hc_i+he_j+b1).W2+b2)-0.5)
// One wave per (b,i) row, lane = channel c (CD == 32 == wave32). Only j<=i written.

__global__ void __launch_bounds__(32)
gmod_kernel(const float* __restrict__ hcb, const float* __restrict__ he,
            const float* __restrict__ W2, const float* __restrict__ b2,
            float* __restrict__ Gmod) {
    const int lane = threadIdx.x;
    const int row  = blockIdx.x;              // b*L + i
    const int b = row >> 9, i = row & (Lsz - 1);

    const float hv  = hcb[(size_t)row * CD + lane];
    const float w2  = W2[lane];
    const float b2v = b2[0];
    const float* heB = he + (size_t)b * Lsz * CD;
    float* Grow = Gmod + ((size_t)b * Lsz + i) * Lsz;

    for (int j = 0; j <= i; ++j) {
        float x = hv + heB[(size_t)j * CD + lane];
        float g = 0.5f * x * (1.0f + erff(x * 0.70710678118f));   // exact gelu
        float t = g * w2;
#pragma unroll
        for (int o = 16; o > 0; o >>= 1) t += __shfl_xor(t, o, 32);
        if (lane == 0) {
            float s = t + b2v;
            Grow[j] = ALPHA * (1.0f / (1.0f + __expf(-s)) - 0.5f);
        }
    }
}

// ---- Fused causal attention per (b, h, 16-row query tile) ------------------
// scores = (Q K^T)/8 + Gmod + causal mask -> softmax -> P(bf16) @ V, all WMMA.

__global__ void __launch_bounds__(32)
attn_kernel(const __bf16* __restrict__ Qb, const __bf16* __restrict__ Kb,
            const __bf16* __restrict__ Vb, const float* __restrict__ Gmod,
            __bf16* __restrict__ Ob) {
    __shared__ float  sS[16][Lsz];            // 32 KB
    __shared__ __bf16 sP[16][Lsz];            // 16 KB

    const int lane = threadIdx.x & 31;
    const int half = lane >> 4, lx = lane & 15;
    const int bh = blockIdx.x;
    const int b = bh >> 4, h = bh & 15;
    const int itile = blockIdx.y, i0 = itile * 16;
    const int rowBase = b * Lsz;
    const int hd0 = h * HD;

    // Q tile (16 x 64) as two A fragments
    v16bf q0 = load_a_rm(Qb, rowBase + i0, hd0,      Dsz);
    v16bf q1 = load_a_rm(Qb, rowBase + i0, hd0 + 32, Dsz);

    // ---- scores tiles: S = Q K^T / sqrt(HD) + Gmod, causal masked ----
    for (int jt = 0; jt <= itile; ++jt) {
        const int j0 = jt * 16;
        v16bf kf0 = load_b_nt(Kb, rowBase + j0, hd0,      Dsz);
        v16bf kf1 = load_b_nt(Kb, rowBase + j0, hd0 + 32, Dsz);
        v8f s = {};
        s = WMMA_BF16(q0, kf0, s);
        s = WMMA_BF16(q1, kf1, s);

        const int j = j0 + lx;
#pragma unroll
        for (int r = 0; r < 8; ++r) {
            const int m = r + 8 * half;
            const int i = i0 + m;
            float val = -NEG_BIG;
            if (j <= i)
                val = s[r] * 0.125f + Gmod[((size_t)b * Lsz + i) * Lsz + j];
            sS[m][j] = val;
        }
    }
    __syncthreads();

    // ---- row softmax over j in [0, nj) ----
    const int nj = (itile + 1) * 16;
    for (int m = 0; m < 16; ++m) {
        float mx = -NEG_BIG;
        for (int j = lane; j < nj; j += 32) mx = fmaxf(mx, sS[m][j]);
#pragma unroll
        for (int o = 16; o > 0; o >>= 1) mx = fmaxf(mx, __shfl_xor(mx, o, 32));
        float sum = 0.0f;
        for (int j = lane; j < nj; j += 32) {
            float e = __expf(sS[m][j] - mx);
            sS[m][j] = e;
            sum += e;
        }
#pragma unroll
        for (int o = 16; o > 0; o >>= 1) sum += __shfl_xor(sum, o, 32);
        const float inv = 1.0f / sum;
        for (int j = lane; j < nj; j += 32) sP[m][j] = (__bf16)(sS[m][j] * inv);
    }
    // zero-pad P to a K=32 boundary for the WMMA contraction
    const int nj32 = (nj + 31) & ~31;
    for (int m = 0; m < 16; ++m)
        for (int j = nj + lane; j < nj32; j += 32) sP[m][j] = (__bf16)0.0f;
    __syncthreads();

    // ---- O tile = P @ V (16 x 64) ----
    v8f o0 = {}, o1 = {}, o2 = {}, o3 = {};
    const __bf16* P = &sP[0][0];
    for (int kk = 0; kk < nj32; kk += 32) {
        v16bf ap = load_a_rm(P, 0, kk, Lsz);
        v16bf v0 = load_b_rm(Vb, rowBase + kk, hd0 +  0, Dsz);
        v16bf v1 = load_b_rm(Vb, rowBase + kk, hd0 + 16, Dsz);
        v16bf v2 = load_b_rm(Vb, rowBase + kk, hd0 + 32, Dsz);
        v16bf v3 = load_b_rm(Vb, rowBase + kk, hd0 + 48, Dsz);
        o0 = WMMA_BF16(ap, v0, o0);
        o1 = WMMA_BF16(ap, v1, o1);
        o2 = WMMA_BF16(ap, v2, o2);
        o3 = WMMA_BF16(ap, v3, o3);
    }

#pragma unroll
    for (int r = 0; r < 8; ++r) {
        const size_t row = (size_t)rowBase + i0 + r + 8 * half;
        __bf16* op = Ob + row * Dsz + hd0;
        op[ 0 + lx] = (__bf16)o0[r];
        op[16 + lx] = (__bf16)o1[r];
        op[32 + lx] = (__bf16)o2[r];
        op[48 + lx] = (__bf16)o3[r];
    }
}

// ---------------------------------------------------------------------------

extern "C" void kernel_launch(void* const* d_in, const int* in_sizes, int n_in,
                              void* d_out, int out_size, void* d_ws, size_t ws_size,
                              hipStream_t stream) {
    const float* x   = (const float*)d_in[0];
    const float* Wq  = (const float*)d_in[1];
    const float* Wk  = (const float*)d_in[2];
    const float* Wv  = (const float*)d_in[3];
    const float* Wo  = (const float*)d_in[4];
    const float* Wc  = (const float*)d_in[5];
    const float* We  = (const float*)d_in[6];
    const float* W1c = (const float*)d_in[7];
    const float* W1e = (const float*)d_in[8];
    const float* b1  = (const float*)d_in[9];
    const float* W2  = (const float*)d_in[10];
    const float* b2  = (const float*)d_in[11];
    float* out = (float*)d_out;

    const int BL = Bsz * Lsz;          // 2048
    char* ws = (char*)d_ws;
    size_t off = 0;
    auto alloc = [&](size_t bytes) -> void* {
        void* p = ws + off;
        off = (off + bytes + 255) & ~(size_t)255;
        return p;
    };

    __bf16* xb   = (__bf16*)alloc((size_t)BL * Dsz * 2);
    __bf16* WqT  = (__bf16*)alloc((size_t)Dsz * Dsz * 2);
    __bf16* WkT  = (__bf16*)alloc((size_t)Dsz * Dsz * 2);
    __bf16* WvT  = (__bf16*)alloc((size_t)Dsz * Dsz * 2);
    __bf16* WoT  = (__bf16*)alloc((size_t)Dsz * Dsz * 2);
    __bf16* WcT  = (__bf16*)alloc((size_t)CD * Dsz * 2);
    __bf16* WeT  = (__bf16*)alloc((size_t)CD * Dsz * 2);
    __bf16* Qb   = (__bf16*)alloc((size_t)BL * Dsz * 2);
    __bf16* Kb   = (__bf16*)alloc((size_t)BL * Dsz * 2);
    __bf16* Vb   = (__bf16*)alloc((size_t)BL * Dsz * 2);
    float*  cause  = (float*)alloc((size_t)BL * CD * 4);
    float*  effect = (float*)alloc((size_t)BL * CD * 4);
    float*  hcb    = (float*)alloc((size_t)BL * CD * 4);
    float*  hev    = (float*)alloc((size_t)BL * CD * 4);
    float*  Gmod   = (float*)alloc((size_t)Bsz * Lsz * Lsz * 4);
    __bf16* Ob   = (__bf16*)alloc((size_t)BL * Dsz * 2);
    (void)ws_size; (void)n_in; (void)in_sizes; (void)out_size;

    // 1) precision conversion / weight transposes
    {
        int n = BL * Dsz;
        to_bf16_kernel<<<(n + 255) / 256, 256, 0, stream>>>(x, xb, n);
        int nw = Dsz * Dsz;
        transpose_bf16_kernel<<<(nw + 255) / 256, 256, 0, stream>>>(Wq, WqT, Dsz, Dsz);
        transpose_bf16_kernel<<<(nw + 255) / 256, 256, 0, stream>>>(Wk, WkT, Dsz, Dsz);
        transpose_bf16_kernel<<<(nw + 255) / 256, 256, 0, stream>>>(Wv, WvT, Dsz, Dsz);
        transpose_bf16_kernel<<<(nw + 255) / 256, 256, 0, stream>>>(Wo, WoT, Dsz, Dsz);
        int nc = Dsz * CD;
        transpose_bf16_kernel<<<(nc + 255) / 256, 256, 0, stream>>>(Wc, WcT, Dsz, CD);
        transpose_bf16_kernel<<<(nc + 255) / 256, 256, 0, stream>>>(We, WeT, Dsz, CD);
    }

    // 2) projections (blocked WMMA + async LDS staging)
    {
        dim3 grid(Dsz / 128, BL / 32);         // (8, 64)
        gemm_bf16_wmma_blocked<<<grid, 128, 0, stream>>>(xb, WqT, Qb, BL, Dsz, Dsz, 1);
        gemm_bf16_wmma_blocked<<<grid, 128, 0, stream>>>(xb, WkT, Kb, BL, Dsz, Dsz, 1);
        gemm_bf16_wmma_blocked<<<grid, 128, 0, stream>>>(xb, WvT, Vb, BL, Dsz, Dsz, 1);
        dim3 gridc(1, BL / 16);                // N=32 -> simple kernel, 2 active waves
        gemm_bf16_wmma<<<gridc, 128, 0, stream>>>(xb, WcT, cause,  BL, CD, Dsz, 0);
        gemm_bf16_wmma<<<gridc, 128, 0, stream>>>(xb, WeT, effect, BL, CD, Dsz, 0);
    }

    // 3) tiny MLP projections: hcb = cause@W1c + b1 ; hev = effect@W1e
    {
        int n = BL * CD;
        mlp_proj_kernel<<<(n + 255) / 256, 256, 0, stream>>>(cause,  W1c, b1, hcb, 1);
        mlp_proj_kernel<<<(n + 255) / 256, 256, 0, stream>>>(effect, W1e, b1, hev, 0);
    }

    // 4) pairwise causal bias (lower triangle only)
    gmod_kernel<<<BL, 32, 0, stream>>>(hcb, hev, W2, b2, Gmod);

    // 5) fused causal attention (WMMA QK^T, softmax, WMMA P@V)
    {
        dim3 grid(Bsz * Hn, Lsz / 16);         // (64, 32)
        attn_kernel<<<grid, 32, 0, stream>>>(Qb, Kb, Vb, Gmod, Ob);
    }

    // 6) output projection -> f32 result
    {
        dim3 grid(Dsz / 128, BL / 32);
        gemm_bf16_wmma_blocked<<<grid, 128, 0, stream>>>(Ob, WoT, out, BL, Dsz, Dsz, 0);
    }
}
